// MultiHeadSelfAttention_75806172774916
// MI455X (gfx1250) — matrix operand order repaired and checked by reading, not compiled, not run on hardware
//
#include <hip/hip_runtime.h>
#include <hip/hip_bf16.h>

// ---------------------------------------------------------------------------
// MHSA forward for MI455X (gfx1250): bf16 WMMA (16x16x32) everywhere,
// f32 accumulation, flash-attention with online softmax.
// Round 3: copy-free ping-pong K-loop (2x unroll) in the projection GEMMs to
// eliminate the v_dual_mov rotation + WMMA hazard NOPs seen in round 2.
// ---------------------------------------------------------------------------

#define BATCH   2
#define SEQ     2048
#define DMODEL  1024
#define NHEADS  16
#define DK      64
#define MROWS   (BATCH * SEQ)          // 4096
#define NQKV    (3 * DMODEL)           // 3072

typedef __bf16 bf16_t;
typedef bf16_t v4bf  __attribute__((ext_vector_type(4)));
typedef bf16_t v8bf  __attribute__((ext_vector_type(8)));
typedef bf16_t v16bf __attribute__((ext_vector_type(16)));
typedef float  v4f   __attribute__((ext_vector_type(4)));
typedef float  v8f   __attribute__((ext_vector_type(8)));

union ABfrag { v16bf v; v8bf h[2]; };

// A-fragment (16x32 bf16, M x K), source row-major [rows, ld], at (row0,k0)=base.
// ISA layout: lanes 0-15 row=lane, elems 0..7 -> K 0..7, elems 8..15 -> K 16..23;
//             lanes 16-31 same rows, K offset +8.
__device__ __forceinline__ v16bf load_frag_a(const bf16_t* base, int ld, int lane) {
    const int r  = lane & 15;
    const int hi = lane >> 4;
    const bf16_t* p = base + r * ld + hi * 8;
    ABfrag f;
    f.h[0] = *(const v8bf*)(p);
    f.h[1] = *(const v8bf*)(p + 16);
    return f.v;
}

// B-fragment (32x16 bf16, K x N) from memory stored as [N][K] row-major (ld = K stride).
// ISA layout: lane = column N, lanes 0-15 elems 0..7 -> K 0..7, elems 8..15 -> K 16..23; lanes 16-31 K offset +8.
__device__ __forceinline__ v16bf load_frag_b(const bf16_t* base, int ld, int lane) {
    const int r  = lane & 15;
    const int hi = lane >> 4;
    ABfrag f;
    const bf16_t* p = base + r * ld + hi * 8;
    f.h[0] = *(const v8bf*)(p);
    f.h[1] = *(const v8bf*)(p + 16);
    return f.v;
}

__device__ __forceinline__ v8f wmma_bf16(v16bf a, v16bf b, v8f c) {
    return __builtin_amdgcn_wmma_f32_16x16x32_bf16(
        /*neg_a=*/false, a, /*neg_b=*/false, b,
        /*c_mod=*/(short)0, c, /*reuse_a=*/false, /*reuse_b=*/false);
}

#define VZERO ((v8f){0.f,0.f,0.f,0.f,0.f,0.f,0.f,0.f})

// Load a full 32x64-tile fragment group (2 A + 4 B) at K offset kk.
__device__ __forceinline__ void load_grp(const bf16_t* Abase, const bf16_t* Bbase,
                                         int kk, int lane,
                                         v16bf (&A)[2], v16bf (&B)[4]) {
#pragma unroll
    for (int mi = 0; mi < 2; ++mi)
        A[mi] = load_frag_a(Abase + mi * 16 * DMODEL + kk, DMODEL, lane);
#pragma unroll
    for (int j = 0; j < 4; ++j)
        B[j] = load_frag_b(Bbase + j * 16 * DMODEL + kk, DMODEL, lane);
}

// 8 WMMAs for a 32x64 tile.
__device__ __forceinline__ void wmma_grp(v8f (&acc)[2][4],
                                         const v16bf (&A)[2], const v16bf (&B)[4]) {
#pragma unroll
    for (int mi = 0; mi < 2; ++mi)
#pragma unroll
        for (int j = 0; j < 4; ++j)
            acc[mi][j] = wmma_bf16(A[mi], B[j], acc[mi][j]);
}

// --------------------------------------------------------------------------
// Kernel 0: fp32 -> bf16 conversion, 4-wide (b128 load / b64 store)
// --------------------------------------------------------------------------
__global__ __launch_bounds__(256) void cvt_f32_bf16_kernel(
        const float* __restrict__ in, bf16_t* __restrict__ out, int n4) {
    for (int i = blockIdx.x * blockDim.x + threadIdx.x; i < n4;
         i += gridDim.x * blockDim.x) {
        v4f x = *(const v4f*)(in + 4 * i);
        v4bf y = { (bf16_t)x.x, (bf16_t)x.y, (bf16_t)x.z, (bf16_t)x.w };
        *(v4bf*)(out + 4 * i) = y;
    }
}

// --------------------------------------------------------------------------
// Kernel 1: fused QKV GEMM + RoPE + scatter.
// One wave: 32(row) x 64(col) tile == one full (t, head) slice for 32 tokens.
// Copy-free ping-pong K loop (2x unrolled): compute buf0 @k, refill buf0
// @k+64, compute buf1 @k+32, refill buf1 @k+96.
// Q,K (RoPE'd) -> [B,H,S,64] bf16; V -> transposed [B,H,64,S] bf16.
// --------------------------------------------------------------------------
__global__ __launch_bounds__(256) void qkv_rope_kernel(
        const bf16_t* __restrict__ Xbf, const bf16_t* __restrict__ Wbf,
        const int* __restrict__ tpos,
        bf16_t* __restrict__ Qr, bf16_t* __restrict__ Kr, bf16_t* __restrict__ Vt) {
    const int lane = threadIdx.x & 31;
    const int wv   = threadIdx.x >> 5;
    const int gw   = blockIdx.x * (blockDim.x >> 5) + wv;
    if (gw >= (MROWS / 32) * (NQKV / 64)) return;   // wave-uniform
    const int m0 = (gw / (NQKV / 64)) * 32;
    const int chunk = gw % (NQKV / 64);
    const int n0 = chunk * 64;

    const bf16_t* Abase = Xbf + m0 * DMODEL;
    const bf16_t* Bbase = Wbf + n0 * DMODEL;

    v8f acc[2][4];
#pragma unroll
    for (int mi = 0; mi < 2; ++mi)
#pragma unroll
        for (int j = 0; j < 4; ++j) acc[mi][j] = VZERO;

    // ---- ping-pong pipelined K loop ----
    v16bf A0[2], B0[4], A1[2], B1[4];
    load_grp(Abase, Bbase, 0,  lane, A0, B0);
    load_grp(Abase, Bbase, 32, lane, A1, B1);
    for (int kk = 0; kk < DMODEL - 64; kk += 64) {
        wmma_grp(acc, A0, B0);
        load_grp(Abase, Bbase, kk + 64, lane, A0, B0);
        wmma_grp(acc, A1, B1);
        load_grp(Abase, Bbase, kk + 96, lane, A1, B1);
    }
    wmma_grp(acc, A0, B0);   // k = DMODEL-64
    wmma_grp(acc, A1, B1);   // k = DMODEL-32

    // ---- RoPE + scatter epilogue ----
    const int t  = chunk / NHEADS;        // 0=q, 1=k, 2=v
    const int h  = chunk % NHEADS;
    const int hi = lane >> 4;
    const int c15 = lane & 15;
    const float kLn1e4_over_dk = 9.2103403719761836f / (float)DK;  // ln(1e4)/64

#pragma unroll
    for (int mi = 0; mi < 2; ++mi) {
#pragma unroll
        for (int j = 0; j < 4; ++j) {
            const int ncol = j * 16 + c15;    // 0..63 within head
#pragma unroll
            for (int v = 0; v < 8; ++v) {
                const int g  = m0 + mi * 16 + v + hi * 8;   // global row
                const int b_ = g >> 11;                     // / SEQ
                const int s  = g & (SEQ - 1);
                float val = acc[mi][j][v];
                if (t < 2) {
                    // interleaved RoPE pair: even col x0, odd col x1
                    float partner = __shfl_xor(val, 1, 32);
                    float pos  = (float)tpos[s];
                    float ang  = pos * __expf(-(float)(ncol & ~1) * kLn1e4_over_dk);
                    float c = __cosf(ang), sn = __sinf(ang);
                    val = (ncol & 1) ? (sn * partner + c * val)
                                     : (c * val - sn * partner);
                }
                bf16_t r = (bf16_t)val;
                const int bh = b_ * NHEADS + h;
                if (t == 0)      Qr[(bh * SEQ + s) * DK + ncol] = r;
                else if (t == 1) Kr[(bh * SEQ + s) * DK + ncol] = r;
                else             Vt[(bh * DK + ncol) * SEQ + s] = r;
            }
        }
    }
}

// --------------------------------------------------------------------------
// Kernel 2: flash attention. block = 128 thr (4 waves); block owns (b,h) and
// 64 query rows; each wave owns 16 query rows and iterates key tiles of 32.
// All K and V fragments for a tile load up front (V is in flight through the
// S-WMMAs and softmax VALU work); next tile is prefetched into cache.
// P (16x32) goes C-layout -> LDS -> A-layout for the PV WMMA.
// --------------------------------------------------------------------------
__global__ __launch_bounds__(128) void flash_attn_kernel(
        const bf16_t* __restrict__ Qr, const bf16_t* __restrict__ Kr,
        const bf16_t* __restrict__ Vt, bf16_t* __restrict__ Att) {
    __shared__ bf16_t Pl[4][16 * 32];

    const int lane = threadIdx.x & 31;
    const int wv   = threadIdx.x >> 5;
    const int bh   = blockIdx.x >> 5;          // / (SEQ/64)
    const int qblk = blockIdx.x & 31;
    const int b_   = bh >> 4, h = bh & 15;
    const int q0   = qblk * 64 + wv * 16;
    const int hi   = lane >> 4;
    const int c15  = lane & 15;

    const bf16_t* Qbase = Qr + (bh * SEQ + q0) * DK;
    v16bf aq0 = load_frag_a(Qbase, DK, lane);        // d 0..31
    v16bf aq1 = load_frag_a(Qbase + 32, DK, lane);   // d 32..63

    v8f o[4];
#pragma unroll
    for (int j = 0; j < 4; ++j) o[j] = VZERO;
    float mrow[8], lrow[8];
#pragma unroll
    for (int v = 0; v < 8; ++v) { mrow[v] = -1e30f; lrow[v] = 0.f; }
    const float scale = 0.125f;                      // 1/sqrt(64)

    for (int j0 = 0; j0 <= q0 + 15; j0 += 32) {      // causal key-tile loop
        // ---- issue all K and V fragment loads for this tile up front ----
        v16bf bk[4], bv[4];
#pragma unroll
        for (int n = 0; n < 2; ++n) {
            const bf16_t* Kb = Kr + (bh * SEQ + j0 + n * 16) * DK;
            bk[2 * n]     = load_frag_b(Kb, DK, lane);       // d 0..31
            bk[2 * n + 1] = load_frag_b(Kb + 32, DK, lane);  // d 32..63
        }
#pragma unroll
        for (int jj = 0; jj < 4; ++jj)
            bv[jj] = load_frag_b(Vt + (bh * DK + jj * 16) * SEQ + j0, SEQ, lane);

        // warm the cache for the next key tile while this one computes
        if (j0 + 32 <= q0 + 15) {
            __builtin_prefetch(Kr + (bh * SEQ + j0 + 32 + (lane & 31)) * DK, 0, 1);
            __builtin_prefetch(Vt + (bh * DK + (lane & 31)) * SEQ + j0 + 32, 0, 1);
        }

        // ---- S = Q K^T ----
        v8f sf[2];
#pragma unroll
        for (int n = 0; n < 2; ++n) {
            v8f s = VZERO;
            s = wmma_bf16(aq0, bk[2 * n], s);
            s = wmma_bf16(aq1, bk[2 * n + 1], s);
            sf[n] = s;
        }

        // ---- scale + causal mask + online softmax (row = 16 lanes) ----
        float newm[8];
#pragma unroll
        for (int v = 0; v < 8; ++v) {
            const int q = q0 + v + hi * 8;
            float s0 = sf[0][v] * scale;
            float s1 = sf[1][v] * scale;
            if (j0 + c15 > q)      s0 = -1e30f;
            if (j0 + 16 + c15 > q) s1 = -1e30f;
            sf[0][v] = s0; sf[1][v] = s1;
            float mx = fmaxf(s0, s1);
            mx = fmaxf(mx, __shfl_xor(mx, 1, 32));
            mx = fmaxf(mx, __shfl_xor(mx, 2, 32));
            mx = fmaxf(mx, __shfl_xor(mx, 4, 32));
            mx = fmaxf(mx, __shfl_xor(mx, 8, 32));
            newm[v] = fmaxf(mrow[v], mx);
        }
#pragma unroll
        for (int v = 0; v < 8; ++v) {
            float e0 = __expf(sf[0][v] - newm[v]);
            float e1 = __expf(sf[1][v] - newm[v]);
            float rs = e0 + e1;
            rs += __shfl_xor(rs, 1, 32);
            rs += __shfl_xor(rs, 2, 32);
            rs += __shfl_xor(rs, 4, 32);
            rs += __shfl_xor(rs, 8, 32);
            float alpha = __expf(mrow[v] - newm[v]);
            lrow[v] = lrow[v] * alpha + rs;
            mrow[v] = newm[v];
            o[0][v] *= alpha; o[1][v] *= alpha; o[2][v] *= alpha; o[3][v] *= alpha;
            const int row = v + hi * 8;
            Pl[wv][row * 32 + c15]      = (bf16_t)e0;
            Pl[wv][row * 32 + 16 + c15] = (bf16_t)e1;
        }
        asm volatile("s_wait_dscnt 0x0" ::: "memory");   // wave-private LDS RAW
        v16bf ap = load_frag_a(&Pl[wv][0], 32, lane);    // P as 16x32 A-fragment

        // ---- O += P V (V fragments already resident) ----
#pragma unroll
        for (int jj = 0; jj < 4; ++jj)
            o[jj] = wmma_bf16(ap, bv[jj], o[jj]);
    }

    // normalize, merge heads: Att[b, s, h*64 + d]
#pragma unroll
    for (int v = 0; v < 8; ++v) {
        const int q = q0 + v + hi * 8;
        float invl = 1.0f / lrow[v];
#pragma unroll
        for (int jj = 0; jj < 4; ++jj)
            Att[(b_ * SEQ + q) * DMODEL + h * DK + jj * 16 + c15] =
                (bf16_t)(o[jj][v] * invl);
    }
}

// --------------------------------------------------------------------------
// Kernel 3: output projection. out[m,o] = sum_d wo[o,d] * att[m,d].
// One wave = 32x64 tile, ping-pong K loop, f32 stores to d_out.
// --------------------------------------------------------------------------
__global__ __launch_bounds__(256) void out_proj_kernel(
        const bf16_t* __restrict__ Att, const bf16_t* __restrict__ Wo,
        float* __restrict__ out) {
    const int lane = threadIdx.x & 31;
    const int wv   = threadIdx.x >> 5;
    const int gw   = blockIdx.x * (blockDim.x >> 5) + wv;
    if (gw >= (MROWS / 32) * (DMODEL / 64)) return;
    const int m0 = (gw >> 4) * 32;
    const int n0 = (gw & 15) * 64;

    const bf16_t* Abase = Att + m0 * DMODEL;
    const bf16_t* Bbase = Wo + n0 * DMODEL;

    v8f acc[2][4];
#pragma unroll
    for (int mi = 0; mi < 2; ++mi)
#pragma unroll
        for (int j = 0; j < 4; ++j) acc[mi][j] = VZERO;

    v16bf A0[2], B0[4], A1[2], B1[4];
    load_grp(Abase, Bbase, 0,  lane, A0, B0);
    load_grp(Abase, Bbase, 32, lane, A1, B1);
    for (int kk = 0; kk < DMODEL - 64; kk += 64) {
        wmma_grp(acc, A0, B0);
        load_grp(Abase, Bbase, kk + 64, lane, A0, B0);
        wmma_grp(acc, A1, B1);
        load_grp(Abase, Bbase, kk + 96, lane, A1, B1);
    }
    wmma_grp(acc, A0, B0);   // k = DMODEL-64
    wmma_grp(acc, A1, B1);   // k = DMODEL-32

    const int hi = lane >> 4, c15 = lane & 15;
#pragma unroll
    for (int mi = 0; mi < 2; ++mi)
#pragma unroll
        for (int j = 0; j < 4; ++j)
#pragma unroll
            for (int v = 0; v < 8; ++v)
                out[(m0 + mi * 16 + v + hi * 8) * DMODEL + n0 + j * 16 + c15] =
                    acc[mi][j][v];
}

// --------------------------------------------------------------------------
// Host-side launcher
// --------------------------------------------------------------------------
extern "C" void kernel_launch(void* const* d_in, const int* in_sizes, int n_in,
                              void* d_out, int out_size, void* d_ws, size_t ws_size,
                              hipStream_t stream) {
    const float* x     = (const float*)d_in[0];   // [2,2048,1024]
    const int*   tpos  = (const int*)  d_in[1];   // [2048]
    const float* qkv_w = (const float*)d_in[2];   // [3,16,64,1024] == [3072,1024]
    const float* wo_w  = (const float*)d_in[3];   // [1024,1024]

    char* ws = (char*)d_ws;
    const size_t MB = 1u << 20;
    bf16_t* Xbf  = (bf16_t*)(ws + 0 * MB);   //  8 MB : [4096,1024]
    bf16_t* Wqkv = (bf16_t*)(ws + 8 * MB);   //  6 MB : [3072,1024]
    bf16_t* Wo   = (bf16_t*)(ws + 14 * MB);  //  2 MB : [1024,1024]
    bf16_t* Qr   = (bf16_t*)(ws + 16 * MB);  //  8 MB : [B,H,S,64]
    bf16_t* Kr   = (bf16_t*)(ws + 24 * MB);  //  8 MB : [B,H,S,64]
    bf16_t* Vt   = (bf16_t*)(ws + 32 * MB);  //  8 MB : [B,H,64,S]
    bf16_t* Att  = (bf16_t*)(ws + 40 * MB);  //  8 MB : [B,S,1024]

    cvt_f32_bf16_kernel<<<1024, 256, 0, stream>>>(x, Xbf, MROWS * DMODEL / 4);
    cvt_f32_bf16_kernel<<<1024, 256, 0, stream>>>(qkv_w, Wqkv, NQKV * DMODEL / 4);
    cvt_f32_bf16_kernel<<<512, 256, 0, stream>>>(wo_w, Wo, DMODEL * DMODEL / 4);

    // (4096/32) M-tiles * 48 chunks = 6144 waves = 768 blocks of 8 waves
    qkv_rope_kernel<<<768, 256, 0, stream>>>(Xbf, Wqkv, tpos, Qr, Kr, Vt);

    // B*H*(S/64) = 1024 blocks of 4 waves
    flash_attn_kernel<<<1024, 128, 0, stream>>>(Qr, Kr, Vt, Att);

    // (4096/32) M-tiles * 16 chunks = 2048 waves = 256 blocks of 8 waves
    out_proj_kernel<<<256, 256, 0, stream>>>(Att, Wo, (float*)d_out);
}
